// Head_19413252178463
// MI455X (gfx1250) — matrix operand order repaired and checked by reading, not compiled, not run on hardware
//
#include <hip/hip_runtime.h>

// ---------------------------------------------------------------------------
// Fused single-head causal attention for MI455X (gfx1250, wave32, WMMA).
//   B=1024, T=256, C=H=64.  One workgroup (8 wave32) per batch element.
//   QKV projection + QK^T + causal softmax + PV all fused; K/V/Q live in LDS
//   (f16), scores live in VGPRs.  HBM traffic = read X (64MB) + write O (64MB)
//   -> ~5.5us floor at 23.3 TB/s; matrix work runs on v_wmma_f32_16x16x32_f16.
// ---------------------------------------------------------------------------

typedef _Float16 v16h __attribute__((ext_vector_type(16)));
typedef _Float16 v8h  __attribute__((ext_vector_type(8)));
typedef float    v8f  __attribute__((ext_vector_type(8)));

static __device__ __forceinline__ v8f wmma_f16(v16h a, v16h b, v8f c) {
  // D = A(16x32 f16) * B(32x16 f16) + C(16x16 f32)
  return __builtin_amdgcn_wmma_f32_16x16x32_f16(false, a, false, b,
                                                (short)0, c, false, false);
}

static __device__ __forceinline__ v16h cat8(v8h lo, v8h hi) {
  return __builtin_shufflevector(lo, hi, 0, 1, 2, 3, 4, 5, 6, 7,
                                 8, 9, 10, 11, 12, 13, 14, 15);
}

static __device__ __forceinline__ v8f zero8() {
  v8f z;
#pragma unroll
  for (int i = 0; i < 8; ++i) z[i] = 0.0f;
  return z;
}

// A fragment (16x32, ISA layout: lane m=l&15, half h=l>>4 holds row m,
// K = k0+8h+0..7 in v[0..3] and K = k0+16+8h+0..7 in v[4..7]) built directly
// from f32 global rows (converted to f16 on the fly).
static __device__ __forceinline__ v16h a_frag_from_f32(const float* rowm,
                                                       int k0, int hf) {
  const float* p = rowm + k0 + 8 * hf;
  v8f f0 = *(const v8f*)p;
  v8f f1 = *(const v8f*)(p + 16);
  v8h h0, h1;
#pragma unroll
  for (int i = 0; i < 8; ++i) {
    h0[i] = (_Float16)f0[i];
    h1[i] = (_Float16)f1[i];
  }
  return cat8(h0, h1);
}

// A fragment from an f16 LDS matrix with row stride ld.
static __device__ __forceinline__ v16h a_frag_lds(const _Float16* base,
                                                  int row0, int ld, int k0,
                                                  int lane) {
  const int m = lane & 15, hf = lane >> 4;
  const _Float16* p = base + (row0 + m) * ld + k0 + 8 * hf;
  v8h lo = *(const v8h*)p;
  v8h hi = *(const v8h*)(p + 16);
  return cat8(lo, hi);
}

// B fragment (32x16 KxN) from a *transposed* f16 LDS matrix Bt[n][k]:
// lane n=l&15 (half h=l>>4) needs K = k0+8h+0..7 and k0+16+8h+0..7 of
// column n, i.e. two 16-byte loads.
static __device__ __forceinline__ v16h b_frag_lds(const _Float16* bt, int n0,
                                                  int ld, int k0, int lane) {
  const int n = lane & 15, hf = lane >> 4;
  const _Float16* p = bt + (n0 + n) * ld + k0 + 8 * hf;
  return cat8(*(const v8h*)p, *(const v8h*)(p + 16));
}

__global__ __launch_bounds__(256) void fused_head_kernel(
    const float* __restrict__ X, const float* __restrict__ Wq,
    const float* __restrict__ bq, const float* __restrict__ Wk,
    const float* __restrict__ bk, const float* __restrict__ Wv,
    const float* __restrict__ bv, float* __restrict__ out) {
  // LDS: 24K(W) + 0.75K(bias) + 32K(Q) + 32K(K) + 32K(Vt) + 8K(P) ~= 129KB
  __shared__ __align__(32) _Float16 sW[3 * 64 * 64];  // Wq|Wk|Wv, [h][c] = Bt
  __shared__ float sBias[3 * 64];
  __shared__ __align__(32) _Float16 sQ[256 * 64];   // [t][h], scale folded in
  __shared__ __align__(32) _Float16 sK[256 * 64];   // [s][h]  (= Bt for QK^T)
  __shared__ __align__(32) _Float16 sVt[64 * 256];  // [h][s]  (= Bt for PV)
  __shared__ __align__(32) _Float16 sP[8 * 16 * 32];  // per-wave P staging

  const int tid = threadIdx.x;
  const int lane = tid & 31;
  const int wave = tid >> 5;
  const int hf = lane >> 4;   // lane half (0: lanes 0-15, 1: lanes 16-31)
  const int l15 = lane & 15;
  const int bidx = blockIdx.x;
  const float* Xb = X + (size_t)bidx * (256 * 64);

  // ---- stage weights (f32 -> f16) + biases ----
  {
    const float* ws[3] = {Wq, Wk, Wv};
#pragma unroll
    for (int m = 0; m < 3; ++m)
      for (int i = tid; i < 64 * 64; i += 256)
        sW[m * 4096 + i] = (_Float16)ws[m][i];
    if (tid < 64) {
      sBias[tid] = bq[tid];
      sBias[64 + tid] = bk[tid];
      sBias[128 + tid] = bv[tid];
    }
  }
  __syncthreads();

  // ---- QKV projections -------------------------------------------------
  // Per wave: 2 t-tiles; X A-fragments loaded ONCE per t-tile and reused for
  // all 3 matrices x 4 h-tiles.  `mat` is a compile-time constant after
  // unrolling, so each store path is straight-line (no EXEC divergence).
#pragma unroll
  for (int it = 0; it < 2; ++it) {
    const int tt = wave + 8 * it;  // uniform per wave
    const float* rowm = Xb + (size_t)(16 * tt + l15) * 64;
    v16h a0 = a_frag_from_f32(rowm, 0, hf);
    v16h a1 = a_frag_from_f32(rowm, 32, hf);
    const int t0 = 16 * tt + 8 * hf;  // row of C/D element r is t0 + r
#pragma unroll
    for (int mat = 0; mat < 3; ++mat) {
      const _Float16* wmat = sW + mat * 4096;
#pragma unroll
      for (int hh = 0; hh < 4; ++hh) {
        v16h b0 = b_frag_lds(wmat, hh * 16, 64, 0, lane);
        v16h b1 = b_frag_lds(wmat, hh * 16, 64, 32, lane);
        v8f c = zero8();
        c = wmma_f16(a0, b0, c);
        c = wmma_f16(a1, b1, c);
        const float bias = sBias[mat * 64 + hh * 16 + l15];
        const int h = hh * 16 + l15;
        if (mat == 0) {  // Q: fold 1/sqrt(64); scattered b16 (row in VGPR idx)
#pragma unroll
          for (int r = 0; r < 8; ++r)
            sQ[(t0 + r) * 64 + h] = (_Float16)((c[r] + bias) * 0.125f);
        } else if (mat == 1) {  // K
#pragma unroll
          for (int r = 0; r < 8; ++r)
            sK[(t0 + r) * 64 + h] = (_Float16)(c[r] + bias);
        } else {  // V transposed: consecutive r = consecutive t -> one b128
          v8h hv;
#pragma unroll
          for (int r = 0; r < 8; ++r) hv[r] = (_Float16)(c[r] + bias);
          *(v8h*)&sVt[h * 256 + t0] = hv;
        }
      }
    }
  }
  __syncthreads();

  // ---- attention: wave w owns query tiles {w, 15-w} (balanced causal) ----
  _Float16* pst = sP + wave * (16 * 32);
  const int qts[2] = {wave, 15 - wave};
#pragma unroll
  for (int iq = 0; iq < 2; ++iq) {
    const int qt = qts[iq];
    v16h aq0 = a_frag_lds(sQ, qt * 16, 64, 0, lane);
    v16h aq1 = a_frag_lds(sQ, qt * 16, 64, 32, lane);

    // scores row-block [16 x 256] resident in 16 x v8f accumulators
    v8f acc[16];
#pragma unroll
    for (int n = 0; n < 16; ++n) {
      if (n <= qt) {  // uniform branch: EXEC all-ones at WMMA
        v16h b0 = b_frag_lds(sK, n * 16, 64, 0, lane);
        v16h b1 = b_frag_lds(sK, n * 16, 64, 32, lane);
        v8f c = zero8();
        c = wmma_f16(aq0, b0, c);
        c = wmma_f16(aq1, b1, c);
        if (n == qt) {  // mask inside diagonal tile: col s > row t
#pragma unroll
          for (int r = 0; r < 8; ++r)
            if (l15 > r + 8 * hf) c[r] = -1e9f;
        }
        acc[n] = c;
      } else {
        v8f c;
#pragma unroll
        for (int r = 0; r < 8; ++r) c[r] = -1e9f;
        acc[n] = c;
      }
    }

    // softmax: each row lives in one 16-lane half (col = l15 + 16n)
#pragma unroll
    for (int r = 0; r < 8; ++r) {
      float m = -3.0e38f;
#pragma unroll
      for (int n = 0; n < 16; ++n) m = fmaxf(m, acc[n][r]);
#pragma unroll
      for (int off = 1; off < 16; off <<= 1)
        m = fmaxf(m, __shfl_xor(m, off, 32));
      float s = 0.0f;
#pragma unroll
      for (int n = 0; n < 16; ++n) {
        const float p = __expf(acc[n][r] - m);
        acc[n][r] = p;
        s += p;
      }
#pragma unroll
      for (int off = 1; off < 16; off <<= 1) s += __shfl_xor(s, off, 32);
      const float rinv = 1.0f / s;
#pragma unroll
      for (int n = 0; n < 16; ++n) acc[n][r] *= rinv;
    }

    // O = P * V, K accumulated in chunks of 32 (C/D -> A relayout via LDS)
    v8f o[4];
#pragma unroll
    for (int j = 0; j < 4; ++j) o[j] = zero8();
#pragma unroll
    for (int kc = 0; kc < 8; ++kc) {
      if (2 * kc <= qt) {  // uniform; chunks fully above diagonal contribute 0
#pragma unroll
        for (int u = 0; u < 2; ++u) {
          const int n = 2 * kc + u;  // tile above diag stores exact 0s
#pragma unroll
          for (int r = 0; r < 8; ++r)
            pst[(r + 8 * hf) * 32 + 16 * u + l15] = (_Float16)acc[n][r];
        }
        v16h ap = a_frag_lds(pst, 0, 32, 0, lane);  // same-wave DS: in-order
#pragma unroll
        for (int j = 0; j < 4; ++j) {
          v16h bvf = b_frag_lds(sVt, j * 16, 256, 32 * kc, lane);
          o[j] = wmma_f16(ap, bvf, o[j]);
        }
      }
    }

    // store O tile (f32, 64B contiguous per 16-lane half per VGPR row)
#pragma unroll
    for (int j = 0; j < 4; ++j) {
#pragma unroll
      for (int r = 0; r < 8; ++r) {
        const size_t t = (size_t)bidx * 256 + qt * 16 + r + 8 * hf;
        out[t * 64 + j * 16 + l15] = o[j][r];
      }
    }
  }
}

extern "C" void kernel_launch(void* const* d_in, const int* in_sizes, int n_in,
                              void* d_out, int out_size, void* d_ws,
                              size_t ws_size, hipStream_t stream) {
  const float* X = (const float*)d_in[0];
  const float* Wq = (const float*)d_in[1];
  const float* bq = (const float*)d_in[2];
  const float* Wk = (const float*)d_in[3];
  const float* bk = (const float*)d_in[4];
  const float* Wv = (const float*)d_in[5];
  const float* bv = (const float*)d_in[6];
  float* out = (float*)d_out;
  const int B = in_sizes[0] / (256 * 64);  // 1024
  fused_head_kernel<<<dim3(B), dim3(256), 0, stream>>>(X, Wq, bq, Wk, bk, Wv,
                                                       bv, out);
  (void)n_in;
  (void)out_size;
  (void)d_ws;
  (void)ws_size;
}
